// GaussianRasterizer_81793357185743
// MI455X (gfx1250) — matrix operand order repaired and checked by reading, not compiled, not run on hardware
//
#include <hip/hip_runtime.h>
#include <hip/hip_bf16.h>
#include <stdint.h>

// ---------------- problem constants ----------------
#define HH 256
#define WW 256
#define NN 512
#define CHUNK 32
#define NCHUNK (NN / CHUNK)
#define TANFOV 0.7f
#define ALPHA_MIN (1.0f / 255.0f)

typedef __attribute__((ext_vector_type(16))) _Float16 v16h;
typedef __attribute__((ext_vector_type(8)))  float    v8f;
typedef __attribute__((ext_vector_type(4)))  uint32_t u32x4;
typedef __attribute__((ext_vector_type(8)))  int      i32x8;
typedef __attribute__((ext_vector_type(4)))  int      i32x4;

// output slice offsets (in 4-byte elements of d_out)
#define OFF_COLOR   0
#define OFF_RADII   (HH * WW * 3)            // 196608, 512 int32
#define OFF_OBSERVE (OFF_RADII + NN)         // 197120, 65536 int32
#define OFF_DEPTH   (OFF_OBSERVE + HH * WW)  // 262656
#define OFF_APPOPA  (OFF_DEPTH + HH * WW)    // 328192
#define OFF_CALPHA  (OFF_APPOPA + HH * WW)   // 393728

#if defined(__has_builtin)
#if __has_builtin(__builtin_amdgcn_tensor_load_to_lds) && __has_builtin(__builtin_amdgcn_s_wait_tensorcnt)
#define USE_TDM 1
#endif
#endif

// ============================================================
// Kernel 1: per-gaussian preprocess (project cov3d -> conic2d)
// ============================================================
__global__ void gs_preprocess(const float* __restrict__ means,
                              const float* __restrict__ opac,
                              const float* __restrict__ cols,
                              const float* __restrict__ scales,
                              const float* __restrict__ rots,
                              const float* __restrict__ vm,
                              float* __restrict__ P0,
                              int* __restrict__ radii_out) {
  int i = blockIdx.x * blockDim.x + threadIdx.x;
  if (i >= NN) return;
  const float fx = WW / (2.0f * TANFOV);
  const float fy = HH / (2.0f * TANFOV);
  float m0 = means[i * 3 + 0], m1 = means[i * 3 + 1], m2 = means[i * 3 + 2];
  float R00 = vm[0], R01 = vm[1], R02 = vm[2],  t0 = vm[3];
  float R10 = vm[4], R11 = vm[5], R12 = vm[6],  t1 = vm[7];
  float R20 = vm[8], R21 = vm[9], R22 = vm[10], t2 = vm[11];
  float x = R00 * m0 + R01 * m1 + R02 * m2 + t0;
  float y = R10 * m0 + R11 * m1 + R12 * m2 + t1;
  float z = R20 * m0 + R21 * m1 + R22 * m2 + t2;
  float zc = fmaxf(z, 1e-4f);
  float lim = 1.3f * TANFOV;
  float tx = fminf(fmaxf(x / zc, -lim), lim) * zc;
  float ty = fminf(fmaxf(y / zc, -lim), lim) * zc;
  // quaternion -> rotation
  float qr = rots[i * 4 + 0], qx = rots[i * 4 + 1], qy = rots[i * 4 + 2], qz = rots[i * 4 + 3];
  float qn = rsqrtf(qr * qr + qx * qx + qy * qy + qz * qz);
  qr *= qn; qx *= qn; qy *= qn; qz *= qn;
  float Q00 = 1.f - 2.f * (qy * qy + qz * qz), Q01 = 2.f * (qx * qy - qr * qz), Q02 = 2.f * (qx * qz + qr * qy);
  float Q10 = 2.f * (qx * qy + qr * qz), Q11 = 1.f - 2.f * (qx * qx + qz * qz), Q12 = 2.f * (qy * qz - qr * qx);
  float Q20 = 2.f * (qx * qz - qr * qy), Q21 = 2.f * (qy * qz + qr * qx), Q22 = 1.f - 2.f * (qx * qx + qy * qy);
  float s0 = scales[i * 3 + 0], s1 = scales[i * 3 + 1], s2 = scales[i * 3 + 2];
  float s02 = s0 * s0, s12 = s1 * s1, s22 = s2 * s2;
  // cov3 = Q diag(s^2) Q^T
  float C00 = Q00 * Q00 * s02 + Q01 * Q01 * s12 + Q02 * Q02 * s22;
  float C01 = Q00 * Q10 * s02 + Q01 * Q11 * s12 + Q02 * Q12 * s22;
  float C02 = Q00 * Q20 * s02 + Q01 * Q21 * s12 + Q02 * Q22 * s22;
  float C11 = Q10 * Q10 * s02 + Q11 * Q11 * s12 + Q12 * Q12 * s22;
  float C12 = Q10 * Q20 * s02 + Q11 * Q21 * s12 + Q12 * Q22 * s22;
  float C22 = Q20 * Q20 * s02 + Q21 * Q21 * s12 + Q22 * Q22 * s22;
  float J00 = fx / zc, J02 = -fx * tx / (zc * zc);
  float J11 = fy / zc, J12 = -fy * ty / (zc * zc);
  // T = J @ Rw
  float T00 = J00 * R00 + J02 * R20, T01 = J00 * R01 + J02 * R21, T02 = J00 * R02 + J02 * R22;
  float T10 = J11 * R10 + J12 * R20, T11 = J11 * R11 + J12 * R21, T12 = J11 * R12 + J12 * R22;
  // cov2 = T C T^T + 0.3 I
  float u0 = C00 * T00 + C01 * T01 + C02 * T02;
  float u1 = C01 * T00 + C11 * T01 + C12 * T02;
  float u2 = C02 * T00 + C12 * T01 + C22 * T02;
  float v0 = C00 * T10 + C01 * T11 + C02 * T12;
  float v1 = C01 * T10 + C11 * T11 + C12 * T12;
  float v2 = C02 * T10 + C12 * T11 + C22 * T12;
  float a = T00 * u0 + T01 * u1 + T02 * u2 + 0.3f;
  float b = T10 * u0 + T11 * u1 + T12 * u2;
  float c = T10 * v0 + T11 * v1 + T12 * v2 + 0.3f;
  float det = a * c - b * b;
  float invd = 1.0f / fmaxf(det, 1e-12f);
  float ca = c * invd, cb = -b * invd, cc = a * invd;
  float mid = 0.5f * (a + c);
  float lam1 = mid + sqrtf(fmaxf(mid * mid - det, 0.1f));
  radii_out[i] = (int)ceilf(3.0f * sqrtf(lam1));
  float mx = fx * x / zc + 0.5f * WW - 0.5f;
  float my = fy * y / zc + 0.5f * HH - 0.5f;
  float valid = (z > 0.2f && det > 0.0f) ? 1.0f : 0.0f;
  float* o = &P0[i * 12];
  o[0] = mx;  o[1] = my;  o[2] = ca;  o[3] = cb;
  o[4] = cc;  o[5] = opac[i];
  o[6] = cols[i * 3 + 0]; o[7] = cols[i * 3 + 1];
  o[8] = cols[i * 3 + 2]; o[9] = z;  o[10] = valid; o[11] = 0.f;
}

// ============================================================
// Kernel 2: depth sort (bitonic, 512 keys, one block) + scatter
// ============================================================
__global__ void gs_sort_scatter(const float* __restrict__ P0, float* __restrict__ P1) {
  __shared__ float skey[NN];
  __shared__ int   sidx[NN];
  int i = threadIdx.x;
  skey[i] = P0[i * 12 + 9];
  sidx[i] = i;
  __syncthreads();
  for (int sz = 2; sz <= NN; sz <<= 1) {
    for (int j = sz >> 1; j > 0; j >>= 1) {
      int ixj = i ^ j;
      if (ixj > i) {
        bool up = ((i & sz) == 0);
        float ki = skey[i], kj = skey[ixj];
        if ((ki > kj) == up) {
          skey[i] = kj; skey[ixj] = ki;
          int t = sidx[i]; sidx[i] = sidx[ixj]; sidx[ixj] = t;
        }
      }
      __syncthreads();
    }
  }
  int src = sidx[i];
  for (int t = 0; t < 12; t++) P1[i * 12 + t] = P0[src * 12 + t];
}

// ============================================================
// TDM: async global->LDS tile load of one 32-gaussian chunk
// (6-arg builtin form: g0, g1, g2, g3, g4, cpol)
// ============================================================
#ifdef USE_TDM
__device__ __forceinline__ void tdm_load_chunk(const float* P1, int chunk, void* ldsPtr) {
  uint32_t lds = (uint32_t)(uintptr_t)ldsPtr;  // low 32 bits = LDS offset
  uint64_t ga = (uint64_t)(uintptr_t)P1 + (uint64_t)chunk * CHUNK * 12u * 4u;
  // D# group0: count=1, lds_addr, global_addr[56:0], type=2
  u32x4 g0 = { 1u, lds, (uint32_t)ga,
               (uint32_t)((ga >> 32) & 0x01FFFFFFull) | (2u << 30) };
  // D# group1: data_size=2(4B); tensor_dim0=12; tensor_dim1=512;
  //            tile_dim0=12; tile_dim1=32; tensor_dim0_stride=12
  i32x8 g1 = { (int)(2u << 16),        // [17:16] data_size=4B, mask=0, no flags
               (int)(12u << 16),       // tensor_dim0[15:0] in [31:16]
               (int)(512u << 16),      // tensor_dim0[31:16]=0 | tensor_dim1[15:0]
               (int)(12u << 16),       // tensor_dim1[31:16]=0 | tile_dim0=12
               32,                     // tile_dim1=32 | tile_dim2=0
               12,                     // tensor_dim0_stride[31:0]
               0, 0 };                 // stride0 hi | tensor_dim1_stride (unused, 2D)
  i32x4 g2 = {0, 0, 0, 0};
  i32x4 g3 = {0, 0, 0, 0};
  i32x8 g4 = {0, 0, 0, 0, 0, 0, 0, 0};
  __builtin_amdgcn_tensor_load_to_lds(g0, g1, g2, g3, g4, 0);
}
#endif

// ============================================================
// B fragment: pure LDS gather of sorted attrs (wave-uniform data,
// lane-indexed addresses) -- hoisted out of the sequential scan.
// Column nCol<4 holds {r,g,b,z}; rows K = {0-7,16-23} (lane<16), {8-15,24-31} (lane>=16).
// ============================================================
__device__ __forceinline__ void pack_b(const float (*__restrict__ G)[12],
                                       int koffB, int nCol, v16h& bv) {
  if (nCol < 4) {
#pragma unroll
    for (int e = 0; e < 16; e++)
      bv[e] = (_Float16)G[(koffB >> 1) + (e & 7) + ((e & 8) ? 16 : 0)][6 + nCol];
  }
}

// ============================================================
// Sequential transmittance scan over one 32-gaussian chunk.
// Fully unrolled so the A-slot ownership test constant-folds to a
// single loop-invariant lane mask + one cndmask per gaussian.
// A fragment (16x32 f16): lane<16 -> K{0-7,16-23}; lane>=16 -> K{8-15,24-31}
// ============================================================
__device__ __forceinline__ void scan_chunk(const float (*__restrict__ G)[12],
                                           float gxf, float gyf, bool lowHalf,
                                           float& Tcur, float& mw, int& cnt,
                                           v16h& av) {
#pragma unroll
  for (int g = 0; g < CHUNK; g++) {
    const float4* q = (const float4*)G[g];   // 48B/gaussian -> 3x ds_load_b128 (broadcast)
    float4 q0 = q[0], q1 = q[1], q2 = q[2];
    float dx = gxf - q0.x, dy = gyf - q0.y;
    float pw = -0.5f * (q0.z * dx * dx + q1.x * dy * dy) - q0.w * dx * dy;
    float al = fminf(0.99f, q1.y * __expf(fminf(pw, 0.0f)));
    bool keep = (pw <= 0.0f) && (q2.z > 0.5f) && (al >= ALPHA_MIN);
    float aa = keep ? al : 0.0f;
    float w = aa * Tcur;
    mw = fmaxf(mw, w);
    cnt += keep ? 1 : 0;
    Tcur *= (1.0f - aa);
    // static slot mapping (g is a compile-time constant here):
    //   g in [0,8)|[16,24) belongs to lanes 0-15; [8,16)|[24,32) to lanes 16-31
    const bool mineLow = ((g >> 3) & 1) == 0;
    const int slot = (g & 7) | ((g & 16) ? 8 : 0);
    bool mine = mineLow ? lowHalf : !lowHalf;
    av[slot] = mine ? (_Float16)w : av[slot];
  }
}

// ============================================================
// Kernel 3: rasterize. 256 thr = 8 waves; wave = 16 pixels.
// Lane pair (L, L+16) scans one pixel; WMMA accumulates
// [r,g,b,depth] channels on the matrix pipe (co-executes w/ VALU),
// TDM double-buffers gaussian params into LDS.
// ============================================================
__global__ void __launch_bounds__(256) gs_raster(const float* __restrict__ P1,
                                                 const float* __restrict__ bg,
                                                 float* __restrict__ out) {
  __shared__ float sg[2][CHUNK][12];
  __shared__ float sTf[128], sMw[128];
  __shared__ int sCnt[128];

  const int tid = threadIdx.x;
  const int lane = tid & 31, wave = tid >> 5;
  const int waveGlobal = blockIdx.x * 8 + wave;
  const int p = lane & 15;                 // pixel row (M) within wave tile
  const int pix = waveGlobal * 16 + p;     // global flat pixel id
  const float gxf = (float)(pix & (WW - 1));
  const float gyf = (float)(pix >> 8);
  const bool lowHalf = (lane < 16);
  const int nCol = lane & 15;              // B/D column (channel)
  const int koffB = lowHalf ? 0 : 16;      // B-fragment K offset

  float Tcur = 1.0f, mw = 0.0f;
  int cnt = 0;
  v8f acc;
  v16h av, bv;
#pragma unroll
  for (int t = 0; t < 8; t++) acc[t] = 0.0f;
#pragma unroll
  for (int t = 0; t < 16; t++) { av[t] = (_Float16)0.0f; bv[t] = (_Float16)0.0f; }

#ifdef USE_TDM
  if (tid < 32) tdm_load_chunk(P1, 0, &sg[0][0][0]);
  for (int c = 0; c < NCHUNK; c++) {
    int cur = c & 1;
    if (tid < 32) __builtin_amdgcn_s_wait_tensorcnt(0);
    __syncthreads();  // chunk c landed in LDS; chunk c-1 reads done everywhere
    if (tid < 32 && c + 1 < NCHUNK)
      tdm_load_chunk(P1, c + 1, &sg[cur ^ 1][0][0]);  // overlap next DMA w/ compute
    pack_b(sg[cur], koffB, nCol, bv);
    scan_chunk(sg[cur], gxf, gyf, lowHalf, Tcur, mw, cnt, av);
    // D = A(16x32 f16 weights) * B(32x16, cols 0..3 = r,g,b,z) + C
    acc = __builtin_amdgcn_wmma_f32_16x16x32_f16(false, av, false, bv,
                                                 (short)0, acc, false, false);
  }
#else
  for (int c = 0; c < NCHUNK; c++) {
    __syncthreads();
    for (int t = tid; t < CHUNK * 12; t += 256)
      ((float*)sg[0])[t] = P1[c * CHUNK * 12 + t];
    __syncthreads();
    pack_b(sg[0], koffB, nCol, bv);
    scan_chunk(sg[0], gxf, gyf, lowHalf, Tcur, mw, cnt, av);
    acc = __builtin_amdgcn_wmma_f32_16x16x32_f16(false, av, false, bv,
                                                 (short)0, acc, false, false);
  }
#endif

  // exchange per-pixel scalars (owned by lane pair) to channel-writer lanes
  if (lane < 16) {
    int li = wave * 16 + p;
    sTf[li] = Tcur; sMw[li] = mw; sCnt[li] = cnt;
  }
  __syncthreads();

  float* colorO = out + OFF_COLOR;
  int*   obsO   = (int*)(out + OFF_OBSERVE);
  float* depO   = out + OFF_DEPTH;
  float* opaO   = out + OFF_APPOPA;
  float* calO   = out + OFF_CALPHA;

  if (nCol < 4) {
    float bgc = (nCol < 3) ? bg[nCol] : 0.0f;
#pragma unroll
    for (int v = 0; v < 8; v++) {
      int m = v + (lowHalf ? 0 : 8);       // C/D row layout: lanes>=16 hold M=8..15
      int li = wave * 16 + m;
      int pg = waveGlobal * 16 + m;
      float val = acc[v];
      if (nCol < 3) colorO[pg * 3 + nCol] = val + sTf[li] * bgc;
      else          depO[pg] = val;
    }
  }
  if (lane < 16) {
    calO[pix] = 1.0f - Tcur;
    opaO[pix] = mw;
    obsO[pix] = cnt;
  }
}

// ============================================================
extern "C" void kernel_launch(void* const* d_in, const int* in_sizes, int n_in,
                              void* d_out, int out_size, void* d_ws, size_t ws_size,
                              hipStream_t stream) {
  const float* means  = (const float*)d_in[0];
  const float* opac   = (const float*)d_in[1];
  const float* cols   = (const float*)d_in[2];
  const float* scales = (const float*)d_in[3];
  const float* rots   = (const float*)d_in[4];
  const float* bg     = (const float*)d_in[5];
  const float* vm     = (const float*)d_in[6];
  float* out = (float*)d_out;
  float* P0 = (float*)d_ws;          // unsorted params  [512][12]
  float* P1 = P0 + NN * 12;          // depth-sorted params [512][12]
  int* radii = (int*)(out + OFF_RADII);

  gs_preprocess<<<2, 256, 0, stream>>>(means, opac, cols, scales, rots, vm, P0, radii);
  gs_sort_scatter<<<1, NN, 0, stream>>>(P0, P1);
  gs_raster<<<(HH * WW) / 128, 256, 0, stream>>>(P1, bg, out);
}
